// SGC_9569187135830
// MI455X (gfx1250) — compile-verified, matrix operand order checked by reading, not run on hardware
//
#include <hip/hip_runtime.h>

typedef __attribute__((ext_vector_type(16))) _Float16 v16h;
typedef __attribute__((ext_vector_type(8)))  _Float16 v8h;
typedef __attribute__((ext_vector_type(8)))  float    v8f;
typedef __attribute__((ext_vector_type(4)))  float    v4f;

// ---------------- problem constants ----------------
constexpr int Bsz = 32, NB = 10, V1 = 3, V2 = 3, S = 2, J = 25;
constexpr int DIN = 24, H1 = 48, H2 = 72, NFEAT = 256, NCLASS = 60;
constexpr int NROWS  = Bsz * NB * V1 * V2 * S * J;   // 144000
constexpr int NGROUP = NROWS / J;                    // 5760
constexpr int KVEC   = J * NFEAT;                    // 6400
constexpr float EPS = 1e-5f;

// weight tile regions inside d_ws (f16 elements); one tile = 512 f16 (32x16 KxN)
constexpr int W1_KT = 1, W1_NT = 3;    // 24->32 K, 48 N
constexpr int W2_KT = 2, W2_NT = 5;    // 48->64 K, 72->80 N
constexpr int W3_KT = 3, W3_NT = 16;   // 72->96 K, 256 N
constexpr int WC_KT = 200, WC_NT = 4;  // 6400 K,  60->64 N
constexpr int OFF_W1 = 0;
constexpr int OFF_W2 = OFF_W1 + W1_KT * W1_NT * 512;           // 1536
constexpr int OFF_W3 = OFF_W2 + W2_KT * W2_NT * 512;           // 6656
constexpr int OFF_WC = OFF_W3 + W3_KT * W3_NT * 512;           // 31232
constexpr int WBUF_TOTAL = OFF_WC + WC_KT * WC_NT * 512;       // 440832 f16
constexpr size_t WS_H3_OFF     = (size_t)1 << 20;              // h3 (f16)  144000*256
constexpr size_t WS_LOGITS_OFF = (size_t)80 << 20;             // logits f32 5760*64

// One-instruction XOR butterfly within 32 lanes: ds_swizzle_b32,
// group-of-32 mode, lane' = lane ^ xor (and_mask=0x1f, or_mask=0).
template <int XOR>
__device__ inline float xor_swz(float v) {
  constexpr int PAT = (XOR << 10) | 0x1f;
  return __int_as_float(__builtin_amdgcn_ds_swizzle(__float_as_int(v), PAT));
}

// ---------------------------------------------------------------------------
// Prep: swizzle f32 weights into f16 WMMA B-operand tiles.
// Tile (kt,nt): lane L holds column n = nt*16 + L%16, VGPR v holds pair
// (k, k+1) with k = kt*32 + (v%4)*2 + (L/16)*8 + (v/4)*16. Stored so that
// lane L reads its 16 halfwords at tileBase + L*16 (two b128 loads).
// ---------------------------------------------------------------------------
__device__ inline _Float16 wtile_elem(const float* W, int K, int N, int NT,
                                      int tile, int e) {
  int kt = tile / NT, nt = tile % NT;
  int L = e >> 4, rem = e & 15, v = rem >> 1, h = rem & 1;
  int n = nt * 16 + (L & 15);
  int k = kt * 32 + (v & 3) * 2 + (L >> 4) * 8 + (v >> 2) * 16 + h;
  if (k >= K || n >= N) return (_Float16)0.f;
  return (_Float16)W[(size_t)k * N + n];
}

__global__ void prep_weights(const float* __restrict__ W1, const float* __restrict__ W2,
                             const float* __restrict__ W3, const float* __restrict__ Wc,
                             _Float16* __restrict__ wbuf) {
  int tid = blockIdx.x * blockDim.x + threadIdx.x;
  if (tid >= WBUF_TOTAL) return;
  _Float16 r;
  if (tid < OFF_W2) {
    int l = tid - OFF_W1; r = wtile_elem(W1, DIN, H1, W1_NT, l / 512, l % 512);
  } else if (tid < OFF_W3) {
    int l = tid - OFF_W2; r = wtile_elem(W2, H1, H2, W2_NT, l / 512, l % 512);
  } else if (tid < OFF_WC) {
    int l = tid - OFF_W3; r = wtile_elem(W3, H2, NFEAT, W3_NT, l / 512, l % 512);
  } else {
    int l = tid - OFF_WC; r = wtile_elem(Wc, KVEC, NCLASS, WC_NT, l / 512, l % 512);
  }
  wbuf[tid] = r;
}

// ---------------------------------------------------------------------------
// WMMA helpers
// ---------------------------------------------------------------------------
template <int KT, int NT>
__device__ inline void mm_tiles(const v16h* a, const _Float16* __restrict__ wbase,
                                v8f* acc, int lane) {
#pragma unroll
  for (int nt = 0; nt < NT; ++nt) acc[nt] = (v8f){};
#pragma unroll
  for (int kt = 0; kt < KT; ++kt) {
#pragma unroll
    for (int nt = 0; nt < NT; ++nt) {
      v16h b = *(const v16h*)(wbase + (kt * NT + nt) * 512 + lane * 16);
      acc[nt] = __builtin_amdgcn_wmma_f32_16x16x32_f16(
          false, a[kt], false, b, (short)0, acc[nt], false, false);
    }
  }
}

// LayerNorm + affine (+optional relu) in place on C-layout accumulators.
// Lane holds column n = nt*16 + lm; VGPR r holds row r + 8*half. Row stats
// via 4-step XOR butterfly within each 16-lane half (ds_swizzle_b32).
template <int NT>
__device__ inline void ln_affine(v8f* acc, const float* __restrict__ bias,
                                 const float* __restrict__ gam,
                                 const float* __restrict__ bet,
                                 int H, bool do_relu, int lm) {
  float inv_h = 1.0f / (float)H;
  float msk[NT], gg[NT], bb2[NT];
#pragma unroll
  for (int nt = 0; nt < NT; ++nt) {
    int n = nt * 16 + lm;
    bool valid = (n < H);
    msk[nt] = valid ? 1.f : 0.f;
    float bb = valid ? bias[n] : 0.f;
    gg[nt]  = valid ? gam[n] : 0.f;
    bb2[nt] = valid ? bet[n] : 0.f;
#pragma unroll
    for (int r = 0; r < 8; ++r) acc[nt][r] += bb;
  }
#pragma unroll
  for (int r = 0; r < 8; ++r) {
    float s = 0.f, q = 0.f;
#pragma unroll
    for (int nt = 0; nt < NT; ++nt) {
      float v = acc[nt][r] * msk[nt];
      s += v; q += v * v;
    }
    s += xor_swz<1>(s);  q += xor_swz<1>(q);
    s += xor_swz<2>(s);  q += xor_swz<2>(q);
    s += xor_swz<4>(s);  q += xor_swz<4>(q);
    s += xor_swz<8>(s);  q += xor_swz<8>(q);
    float mean = s * inv_h;
    float var  = q * inv_h - mean * mean;
    float rstd = rsqrtf(var + EPS);
#pragma unroll
    for (int nt = 0; nt < NT; ++nt) {
      float v = (acc[nt][r] - mean) * rstd * gg[nt] + bb2[nt];
      if (do_relu) v = fmaxf(v, 0.f);
      acc[nt][r] = v;
    }
  }
}

template <int NT>
__device__ inline void store_stage(_Float16 (*st)[96], const v8f* acc, int H,
                                   int lm, int half) {
#pragma unroll
  for (int nt = 0; nt < NT; ++nt) {
    int n = nt * 16 + lm;
    if (n < H) {
#pragma unroll
      for (int r = 0; r < 8; ++r) st[r + 8 * half][n] = (_Float16)acc[nt][r];
    }
  }
}

// A-operand (16-bit 16x32 layout): lane L = row lm, halfwords are
// k = kt*32 + half*8 + {0..7} then +16 of the same.
template <int KT>
__device__ inline void load_a_stage(const _Float16 (*st)[96], v16h* a,
                                    int lm, int half) {
#pragma unroll
  for (int kt = 0; kt < KT; ++kt) {
    const _Float16* sp = &st[lm][kt * 32 + half * 8];
    v8h lo = *(const v8h*)sp;
    v8h hi = *(const v8h*)(sp + 16);
#pragma unroll
    for (int i = 0; i < 8; ++i) { a[kt][i] = lo[i]; a[kt][8 + i] = hi[i]; }
  }
}

// ---------------------------------------------------------------------------
// Fused MLP: 3 x (GEMM -> LN [-> relu]), 16 rows per wave, 4 waves per block.
// ---------------------------------------------------------------------------
__global__ __launch_bounds__(128)
void mlp_kernel(const float* __restrict__ x,
                const float* b1, const float* g1, const float* be1,
                const float* b2, const float* g2, const float* be2,
                const float* b3, const float* g3, const float* be3,
                const _Float16* __restrict__ wbuf,
                _Float16* __restrict__ h3out) {
  __shared__ _Float16 stage[4][16][96];      // inter-layer staging (K-padded)
  __shared__ _Float16 outstage[4][16][256];  // layer-3 output transpose buffer
  const int wave = threadIdx.x >> 5;
  const int lane = threadIdx.x & 31;
  const int lm = lane & 15, half = lane >> 4;
  const int tile = blockIdx.x * 4 + wave;
  const int rowbase = tile * 16;

  // zero stage so K-pad columns read as 0 forever after
  for (int i = threadIdx.x; i < 4 * 16 * 96; i += 128)
    ((_Float16*)stage)[i] = (_Float16)0.f;
  __syncthreads();

  // ---- layer 1: A from x (row lm, k = half*8+{0..7}, 16+half*8+{0..7}) ----
  v16h a1[1];
  {
    const float* xr = x + (size_t)(rowbase + lm) * DIN;
    float c[16];
    *(v4f*)(c + 0) = *(const v4f*)(xr + half * 8);
    *(v4f*)(c + 4) = *(const v4f*)(xr + half * 8 + 4);
    if (half == 0) {
      *(v4f*)(c + 8)  = *(const v4f*)(xr + 16);
      *(v4f*)(c + 12) = *(const v4f*)(xr + 20);
    } else {
#pragma unroll
      for (int i = 8; i < 16; ++i) c[i] = 0.f;   // k = 24..31 pad
    }
#pragma unroll
    for (int i = 0; i < 16; ++i) a1[0][i] = (_Float16)c[i];
  }
  v8f acc1[W1_NT];
  mm_tiles<W1_KT, W1_NT>(a1, wbuf + OFF_W1, acc1, lane);
  ln_affine<W1_NT>(acc1, b1, g1, be1, H1, true, lm);
  store_stage<W1_NT>(stage[wave], acc1, H1, lm, half);
  __syncthreads();

  // ---- layer 2 ----
  v16h a2[W2_KT];
  load_a_stage<W2_KT>(stage[wave], a2, lm, half);
  __syncthreads();   // reads done before layer-2 output overwrites stage
  v8f acc2[W2_NT];
  mm_tiles<W2_KT, W2_NT>(a2, wbuf + OFF_W2, acc2, lane);
  ln_affine<W2_NT>(acc2, b2, g2, be2, H2, true, lm);
  store_stage<W2_NT>(stage[wave], acc2, H2, lm, half);
  __syncthreads();

  // ---- layer 3 (no relu) ----
  v16h a3[W3_KT];
  load_a_stage<W3_KT>(stage[wave], a3, lm, half);
  v8f acc3[W3_NT];
  mm_tiles<W3_KT, W3_NT>(a3, wbuf + OFF_W3, acc3, lane);
  ln_affine<W3_NT>(acc3, b3, g3, be3, NFEAT, false, lm);
#pragma unroll
  for (int nt = 0; nt < W3_NT; ++nt) {
    int n = nt * 16 + lm;
#pragma unroll
    for (int r = 0; r < 8; ++r)
      outstage[wave][r + 8 * half][n] = (_Float16)acc3[nt][r];
  }
  __syncthreads();
  // coalesced f16 row store
  const _Float16* src = &outstage[wave][0][0];
  _Float16* dst = h3out + (size_t)rowbase * NFEAT;
  for (int i = lane; i < 512; i += 32)
    *(v8h*)(dst + i * 8) = *(const v8h*)(src + i * 8);
}

// ---------------------------------------------------------------------------
// SGC propagation, in place on h3 (f16), f32 math. One block per group.
// Adjacency staged into LDS via gfx1250 async load-to-LDS (ASYNCcnt).
// ---------------------------------------------------------------------------
__global__ __launch_bounds__(256)
void sgc_kernel(const float* __restrict__ adj, const float* __restrict__ alphap,
                const int* __restrict__ degreep, _Float16* __restrict__ h3) {
  __shared__ float sadj[J * J];
  const int g = blockIdx.x, t = threadIdx.x;  // t = feature column
  for (int i = t; i < J * J; i += 256) {
    unsigned int loff = (unsigned int)(uintptr_t)&sadj[i];  // LDS byte offset
    const float* gp = adj + i;
    asm volatile("global_load_async_to_lds_b32 %0, %1, off"
                 :: "v"(loff), "v"(gp) : "memory");
  }
  asm volatile("s_wait_asynccnt 0x0" ::: "memory");
  __syncthreads();
  const float alpha = alphap[0];
  const int degree = degreep[0];
  _Float16* base = h3 + (size_t)g * KVEC + t;
  float f0[J], f[J], acc[J];
#pragma unroll
  for (int j = 0; j < J; ++j) {
    f0[j] = (float)base[j * NFEAT];
    f[j] = f0[j];
    acc[j] = 0.f;
  }
  for (int d = 0; d < degree; ++d) {
    float nf[J];
#pragma unroll
    for (int i = 0; i < J; ++i) {
      float s = 0.f;
#pragma unroll
      for (int j = 0; j < J; ++j) s += sadj[i * J + j] * f[j];
      nf[i] = s;
    }
#pragma unroll
    for (int i = 0; i < J; ++i) {
      f[i] = nf[i];
      acc[i] += (1.f - alpha) * nf[i] + alpha * f0[i];
    }
  }
  const float inv = 1.f / (float)degree;
#pragma unroll
  for (int j = 0; j < J; ++j) base[j * NFEAT] = (_Float16)(acc[j] * inv);
}

// ---------------------------------------------------------------------------
// Classifier: [5760 x 6400] f16 @ prepped Wc tiles -> logits f32 [5760 x 64].
// Split-K: 4 waves per block each cover 50 k-tiles; partials reduced in LDS.
// ---------------------------------------------------------------------------
__global__ __launch_bounds__(128)
void cls_kernel(const _Float16* __restrict__ accbuf,
                const _Float16* __restrict__ wc,
                const float* __restrict__ bc, float* __restrict__ logits) {
  __shared__ float part[4][WC_NT][256];      // [wave][nt][r*32 + lane]
  const int wave = threadIdx.x >> 5;
  const int lane = threadIdx.x & 31;
  const int lm = lane & 15, half = lane >> 4;
  const int tile = blockIdx.x;
  const int kt0 = wave * (WC_KT / 4), kt1 = kt0 + (WC_KT / 4);

  v8f acc[WC_NT];
#pragma unroll
  for (int nt = 0; nt < WC_NT; ++nt) acc[nt] = (v8f){};
  const _Float16* arow = accbuf + (size_t)(tile * 16 + lm) * KVEC;
  for (int kt = kt0; kt < kt1; ++kt) {
    const _Float16* ap = arow + kt * 32 + half * 8;
    if (kt + 1 < kt1) __builtin_prefetch(arow + (kt + 1) * 32, 0, 1);
    v8h lo = *(const v8h*)ap;
    v8h hi = *(const v8h*)(ap + 16);
    v16h a;
#pragma unroll
    for (int i = 0; i < 8; ++i) { a[i] = lo[i]; a[8 + i] = hi[i]; }
#pragma unroll
    for (int nt = 0; nt < WC_NT; ++nt) {
      v16h b = *(const v16h*)(wc + (kt * WC_NT + nt) * 512 + lane * 16);
      acc[nt] = __builtin_amdgcn_wmma_f32_16x16x32_f16(
          false, a, false, b, (short)0, acc[nt], false, false);
    }
  }
#pragma unroll
  for (int nt = 0; nt < WC_NT; ++nt)
#pragma unroll
    for (int r = 0; r < 8; ++r) part[wave][nt][r * 32 + lane] = acc[nt][r];
  __syncthreads();

  // cross-wave reduction + bias + store (1024 outputs, 128 threads)
  for (int e = threadIdx.x; e < WC_NT * 256; e += 128) {
    int nt = e >> 8, rl = e & 255;
    int r = rl >> 5, L = rl & 31;
    float s = part[0][nt][rl] + part[1][nt][rl] + part[2][nt][rl] + part[3][nt][rl];
    int n = nt * 16 + (L & 15);
    int row = tile * 16 + r + 8 * (L >> 4);
    float bb = (n < NCLASS) ? bc[n] : 0.f;
    logits[(size_t)row * 64 + n] = s + bb;
  }
}

// ---------------------------------------------------------------------------
// Reductions
// ---------------------------------------------------------------------------
__global__ void block_mean_kernel(const float* __restrict__ logits,
                                  float* __restrict__ out) {
  int i = blockIdx.x * blockDim.x + threadIdx.x;
  if (i >= (NROWS / J / S) * NCLASS) return;   // 172800
  int c = i % NCLASS, q = i / NCLASS;          // q = (b,nb,v1,v2) flat
  out[i] = 0.5f * (logits[(size_t)(2 * q) * 64 + c] +
                   logits[(size_t)(2 * q + 1) * 64 + c]);
}

__global__ void pred_kernel(const float* __restrict__ bm, float* __restrict__ out) {
  int i = blockIdx.x * blockDim.x + threadIdx.x;
  if (i >= Bsz * NCLASS) return;
  int c = i % NCLASS, b = i / NCLASS;
  float s = 0.f;
  for (int r = 0; r < NB * V1 * V2; ++r)       // 90
    s += bm[(size_t)(b * NB * V1 * V2 + r) * NCLASS + c];
  out[i] = s * (1.f / (NB * V1 * V2));
}

// ---------------------------------------------------------------------------
extern "C" void kernel_launch(void* const* d_in, const int* in_sizes, int n_in,
                              void* d_out, int out_size, void* d_ws, size_t ws_size,
                              hipStream_t stream) {
  const float* x   = (const float*)d_in[0];
  const float* adj = (const float*)d_in[1];
  const float* W1  = (const float*)d_in[2];
  const float* b1  = (const float*)d_in[3];
  const float* g1  = (const float*)d_in[4];
  const float* be1 = (const float*)d_in[5];
  const float* W2  = (const float*)d_in[6];
  const float* b2  = (const float*)d_in[7];
  const float* g2  = (const float*)d_in[8];
  const float* be2 = (const float*)d_in[9];
  const float* W3  = (const float*)d_in[10];
  const float* b3  = (const float*)d_in[11];
  const float* g3  = (const float*)d_in[12];
  const float* be3 = (const float*)d_in[13];
  const float* Wc  = (const float*)d_in[14];
  const float* bc  = (const float*)d_in[15];
  const int* degree  = (const int*)d_in[17];
  const float* alpha = (const float*)d_in[19];

  _Float16* wbuf = (_Float16*)d_ws;
  _Float16* h3   = (_Float16*)((char*)d_ws + WS_H3_OFF);
  float* logits  = (float*)((char*)d_ws + WS_LOGITS_OFF);
  float* out     = (float*)d_out;

  prep_weights<<<(WBUF_TOTAL + 255) / 256, 256, 0, stream>>>(W1, W2, W3, Wc, wbuf);
  mlp_kernel<<<NROWS / 64, 128, 0, stream>>>(x, b1, g1, be1, b2, g2, be2,
                                             b3, g3, be3, wbuf, h3);
  sgc_kernel<<<NGROUP, 256, 0, stream>>>(adj, alpha, degree, h3);
  cls_kernel<<<NGROUP / 16, 128, 0, stream>>>(h3, wbuf + OFF_WC, bc, logits);
  block_mean_kernel<<<(2880 * NCLASS + 255) / 256, 256, 0, stream>>>(logits, out);
  pred_kernel<<<(Bsz * NCLASS + 255) / 256, 256, 0, stream>>>(out, out + 2880 * NCLASS);
}
